// LTAE_80556406604591
// MI455X (gfx1250) — matrix operand order repaired and checked by reading, not compiled, not run on hardware
//
#include <hip/hip_runtime.h>
#include <stdint.h>

typedef __attribute__((ext_vector_type(16))) __bf16 v16bf;
typedef __attribute__((ext_vector_type(8)))  float  v8f;
typedef __attribute__((ext_vector_type(4)))  int    v4i;

#define AS1 __attribute__((address_space(1)))
#define AS3 __attribute__((address_space(3)))

#if __has_builtin(__builtin_amdgcn_global_load_async_to_lds_b128)
#define HAVE_ASYNC_LDS 1
#else
#define HAVE_ASYNC_LDS 0
#endif

__device__ __forceinline__ void wait_async_zero() {
#if __has_builtin(__builtin_amdgcn_s_wait_asynccnt)
    __builtin_amdgcn_s_wait_asynccnt(0);
#else
    asm volatile("s_wait_asynccnt 0" ::: "memory");
#endif
}

struct U8 { unsigned int v[8]; };

// RNE conversion (cold path: weight prep only)
__device__ __forceinline__ unsigned short f2bf_rne(float f) {
    unsigned int u = __builtin_bit_cast(unsigned int, f);
    u += 0x7FFFu + ((u >> 16) & 1u);
    return (unsigned short)(u >> 16);
}
// Truncating conversions (hot paths)
__device__ __forceinline__ unsigned short f2bf_t(float f) {
    return (unsigned short)(__builtin_bit_cast(unsigned int, f) >> 16);
}
__device__ __forceinline__ unsigned int pack_bf16(float a, float b) {
    unsigned int ua = __builtin_bit_cast(unsigned int, a);
    unsigned int ub = __builtin_bit_cast(unsigned int, b);
    return (ub & 0xFFFF0000u) | (ua >> 16);
}
__device__ __forceinline__ float bf2f(unsigned short h) {
    unsigned int u = ((unsigned int)h) << 16;
    return __builtin_bit_cast(float, u);
}

// Build 16x bf16 fragment from two 16B chunks (p0: elems 0..7, p1: elems 8..15)
__device__ __forceinline__ v16bf load_frag(const unsigned short* p0, const unsigned short* p1) {
    uint4 a = *reinterpret_cast<const uint4*>(p0);
    uint4 b = *reinterpret_cast<const uint4*>(p1);
    U8 u;
    u.v[0] = a.x; u.v[1] = a.y; u.v[2] = a.z; u.v[3] = a.w;
    u.v[4] = b.x; u.v[5] = b.y; u.v[6] = b.z; u.v[7] = b.w;
    return __builtin_bit_cast(v16bf, u);
}

#define T_LEN   64
#define C_IN    128
#define D_MODEL 256
#define N_HEAD  16
#define D_KK    8
#define KDIM    128
#define D_OUT   128

// ---------------------------------------------------------------------------
// Kernel 1: weights fp32 -> bf16 (RNE); fold bias+BN into per-channel scale/shift
// ---------------------------------------------------------------------------
__global__ __launch_bounds__(256)
void ltae_prep(const float* __restrict__ Win, const float* __restrict__ Wk,
               const float* __restrict__ Wmlp,
               const float* __restrict__ b_in, const float* __restrict__ g1,
               const float* __restrict__ be1, const float* __restrict__ m1,
               const float* __restrict__ v1,
               const float* __restrict__ b_mlp, const float* __restrict__ g2,
               const float* __restrict__ be2, const float* __restrict__ m2,
               const float* __restrict__ v2,
               unsigned short* __restrict__ win_bf, unsigned short* __restrict__ wk_bf,
               unsigned short* __restrict__ wmlp_bf,
               float* __restrict__ A1, float* __restrict__ C1,
               float* __restrict__ A2, float* __restrict__ C2)
{
    int i = blockIdx.x * 256 + threadIdx.x;
    if (i < 32768) {
        win_bf[i] = f2bf_rne(Win[i]);
    } else if (i < 65536) {
        int j = i - 32768; wk_bf[j] = f2bf_rne(Wk[j]);
    } else if (i < 98304) {
        int j = i - 65536; wmlp_bf[j] = f2bf_rne(Wmlp[j]);
    } else if (i < 98304 + 256) {
        int d = i - 98304;
        float s = g1[d] * rsqrtf(v1[d] + 1e-5f);
        A1[d] = s;
        C1[d] = (b_in[d] - m1[d]) * s + be1[d];
    } else if (i < 98304 + 256 + 128) {
        int d = i - 98304 - 256;
        float s = g2[d] * rsqrtf(v2[d] + 1e-5f);
        A2[d] = s;
        C2[d] = (b_mlp[d] - m2[d]) * s + be2[d];
    }
}

// ---------------------------------------------------------------------------
// Kernel 2: fused inconv(GEMM1)+BN1+ReLU+PE, key-proj(GEMM2), attention, y
// One workgroup (8 x wave32) per batch element.
// ---------------------------------------------------------------------------
__global__ __launch_bounds__(256)
void ltae_main(const float* __restrict__ x, const int* __restrict__ positions,
               const float* __restrict__ pos_table, const float* __restrict__ query,
               const float* __restrict__ bk,
               const unsigned short* __restrict__ win_bf,
               const unsigned short* __restrict__ wk_bf,
               const float* __restrict__ A1, const float* __restrict__ C1,
               float* __restrict__ y_ws)
{
    __shared__ __align__(16) unsigned char smem[54016];
    unsigned short* xb     = (unsigned short*)(smem);           // [64][128] bf16; later k_bf
    unsigned short* encb   = (unsigned short*)(smem + 16384);   // [64][256] bf16 (overlaid by fp32 x stage)
    float*          scores = (float*)(smem + 49152);            // [16][64]
    int*            posrow = (int*)(smem + 53248);              // [64]
    float*          qs     = (float*)(smem + 53504);            // [16][8]

    const int tid  = threadIdx.x;
    const int lane = tid & 31;
    const int wave = tid >> 5;
    const int hi   = lane >> 4;
    const int lo   = lane & 15;
    const int b    = blockIdx.x;

    const float4* xg = reinterpret_cast<const float4*>(x + (size_t)b * T_LEN * C_IN);

#if HAVE_ASYNC_LDS
    // ---- async DMA x[b] fp32 -> LDS (enc region, dead right now), then cvt ----
    {
        float* xf32 = (float*)(smem + 16384);   // [64][128] fp32 = 32 KB
        #pragma unroll
        for (int i = 0; i < 8; ++i) {
            int idx4 = tid + i * 256;
            __builtin_amdgcn_global_load_async_to_lds_b128(
                (AS1 v4i*)(xg + idx4),
                (AS3 v4i*)(xf32 + idx4 * 4),
                0, 0);
        }
        if (tid < T_LEN) posrow[tid] = positions[b * T_LEN + tid] + 100;
        if (tid < N_HEAD * D_KK) qs[tid] = query[tid];
        // warm weight rows in cache while DMA runs
        __builtin_prefetch(win_bf + tid * C_IN, 0, 3);
        if (tid < KDIM) __builtin_prefetch(wk_bf + tid * D_MODEL, 0, 3);
        wait_async_zero();
        __syncthreads();
        const float4* xl = reinterpret_cast<const float4*>(xf32);
        #pragma unroll
        for (int i = 0; i < 8; ++i) {
            int idx4 = tid + i * 256;
            float4 f = xl[idx4];
            *reinterpret_cast<uint2*>(xb + idx4 * 4) =
                make_uint2(pack_bf16(f.x, f.y), pack_bf16(f.z, f.w));
        }
        __syncthreads();
    }
#else
    // ---- fallback: direct global -> bf16 LDS ----
    #pragma unroll
    for (int i = 0; i < 8; ++i) {
        int idx4 = tid + i * 256;
        float4 f = xg[idx4];
        *reinterpret_cast<uint2*>(xb + idx4 * 4) =
            make_uint2(pack_bf16(f.x, f.y), pack_bf16(f.z, f.w));
    }
    if (tid < T_LEN) posrow[tid] = positions[b * T_LEN + tid] + 100;
    if (tid < N_HEAD * D_KK) qs[tid] = query[tid];
    __builtin_prefetch(win_bf + tid * C_IN, 0, 3);
    if (tid < KDIM) __builtin_prefetch(wk_bf + tid * D_MODEL, 0, 3);
    __syncthreads();
#endif

    // ---- GEMM1: enc = relu((x Win^T)*A1 + C1) + PE -> encb bf16 ----
    // wave owns tm = wave>>1 and 8 n-tiles; A fragments hoisted (loaded once)
    {
        const int tm = wave >> 1;
        const int tnbase = (wave & 1) * 8;
        const int m = tm * 16 + lo;
        v16bf afrag[4];
        #pragma unroll
        for (int ks = 0; ks < 4; ++ks) {
            const unsigned short* pa = xb + m * C_IN + ks * 32 + hi * 8;
            afrag[ks] = load_frag(pa, pa + 16);
        }
        // pos_table row pointers for this wave's 8 output rows (fixed across n-tiles)
        const float* ptrow[8];
        #pragma unroll
        for (int r = 0; r < 8; ++r) {
            int t = tm * 16 + hi * 8 + r;
            ptrow[r] = pos_table + (size_t)posrow[t] * D_MODEL;
        }
        for (int j = 0; j < 8; ++j) {
            const int n = (tnbase + j) * 16 + lo;
            v8f acc = {};
            #pragma unroll
            for (int ks = 0; ks < 4; ++ks) {
                const unsigned short* pb = win_bf + n * C_IN + ks * 32 + hi * 8;
                v16bf bfr = load_frag(pb, pb + 16);
                acc = __builtin_amdgcn_wmma_f32_16x16x32_bf16(false, afrag[ks], false, bfr,
                                                              (short)0, acc, false, false);
            }
            float a1 = A1[n], c1 = C1[n];
            #pragma unroll
            for (int r = 0; r < 8; ++r) {
                int t = tm * 16 + hi * 8 + r;
                float e = acc[r] * a1 + c1;
                e = e > 0.f ? e : 0.f;
                e += ptrow[r][n];
                encb[t * D_MODEL + n] = f2bf_t(e);
            }
        }
    }
    __syncthreads();

    // ---- GEMM2: k = enc Wk^T + bk -> k_bf (reuse xb region) [64][128] ----
    // wave owns tn = wave; B (weight) fragments hoisted
    unsigned short* kbf = xb;
    {
        const int n = wave * 16 + lo;
        v16bf bfrag[8];
        #pragma unroll
        for (int ks = 0; ks < 8; ++ks) {
            const unsigned short* pb = wk_bf + n * D_MODEL + ks * 32 + hi * 8;
            bfrag[ks] = load_frag(pb, pb + 16);
        }
        const float bias = bk[n];
        for (int tm = 0; tm < 4; ++tm) {
            const int m = tm * 16 + lo;
            v8f acc = {};
            #pragma unroll
            for (int ks = 0; ks < 8; ++ks) {
                const unsigned short* pa = encb + m * D_MODEL + ks * 32 + hi * 8;
                v16bf afr = load_frag(pa, pa + 16);
                acc = __builtin_amdgcn_wmma_f32_16x16x32_bf16(false, afr, false, bfrag[ks],
                                                              (short)0, acc, false, false);
            }
            #pragma unroll
            for (int r = 0; r < 8; ++r) {
                int t = tm * 16 + hi * 8 + r;
                kbf[t * KDIM + n] = f2bf_t(acc[r] + bias);
            }
        }
    }
    __syncthreads();

    // ---- scores[h][t] = (q_h . k[t,h]) / sqrt(d_k) ----
    for (int p = tid; p < N_HEAD * T_LEN; p += 256) {
        int h = p >> 6, t = p & 63;
        float s = 0.f;
        #pragma unroll
        for (int d = 0; d < D_KK; ++d)
            s += qs[h * D_KK + d] * bf2f(kbf[t * KDIM + h * D_KK + d]);
        scores[h * 64 + t] = s * 0.35355339059327373f;
    }
    __syncthreads();

    // ---- softmax over t per head ----
    if (tid < N_HEAD) {
        float mx = -1e30f;
        for (int t = 0; t < 64; ++t) mx = fmaxf(mx, scores[tid * 64 + t]);
        float sum = 0.f;
        for (int t = 0; t < 64; ++t) {
            float e = __expf(scores[tid * 64 + t] - mx);
            scores[tid * 64 + t] = e;
            sum += e;
        }
        float inv = 1.f / sum;
        for (int t = 0; t < 64; ++t) scores[tid * 64 + t] *= inv;
    }
    __syncthreads();

    // ---- y[d] = sum_t att[h(d)][t] * enc[t][d] ----
    {
        int d = tid;
        int h = d >> 4;
        float acc = 0.f;
        for (int t = 0; t < 64; ++t)
            acc += scores[h * 64 + t] * bf2f(encb[t * D_MODEL + d]);
        y_ws[(size_t)b * D_MODEL + d] = acc;
    }
}

// ---------------------------------------------------------------------------
// Kernel 3: o = relu((y * Wmlp^T)*A2 + C2)  -- M=8192 WMMA GEMM, 64 rows/WG
// ---------------------------------------------------------------------------
__global__ __launch_bounds__(256)
void ltae_mlp(const float* __restrict__ y, const unsigned short* __restrict__ wmlp_bf,
              const float* __restrict__ A2, const float* __restrict__ C2,
              float* __restrict__ out)
{
    __shared__ __align__(16) unsigned short ybf[64 * D_MODEL];  // 32 KB
    const int tid  = threadIdx.x;
    const int lane = tid & 31;
    const int wave = tid >> 5;
    const int hi   = lane >> 4;
    const int lo   = lane & 15;
    const int m0   = blockIdx.x * 64;

    const float4* yg = reinterpret_cast<const float4*>(y + (size_t)m0 * D_MODEL);
    #pragma unroll
    for (int i = 0; i < 16; ++i) {
        int idx4 = tid + i * 256;
        float4 f = yg[idx4];
        *reinterpret_cast<uint2*>(ybf + idx4 * 4) =
            make_uint2(pack_bf16(f.x, f.y), pack_bf16(f.z, f.w));
    }
    if (tid < KDIM) __builtin_prefetch(wmlp_bf + tid * D_MODEL, 0, 3);
    __syncthreads();

    // wave owns tn = wave; weight B fragments hoisted
    {
        const int n = wave * 16 + lo;
        v16bf bfrag[8];
        #pragma unroll
        for (int ks = 0; ks < 8; ++ks) {
            const unsigned short* pb = wmlp_bf + n * D_MODEL + ks * 32 + hi * 8;
            bfrag[ks] = load_frag(pb, pb + 16);
        }
        const float a2 = A2[n], c2 = C2[n];
        for (int tm = 0; tm < 4; ++tm) {
            const int m = tm * 16 + lo;
            v8f acc = {};
            #pragma unroll
            for (int ks = 0; ks < 8; ++ks) {
                const unsigned short* pa = ybf + m * D_MODEL + ks * 32 + hi * 8;
                v16bf afr = load_frag(pa, pa + 16);
                acc = __builtin_amdgcn_wmma_f32_16x16x32_bf16(false, afr, false, bfrag[ks],
                                                              (short)0, acc, false, false);
            }
            #pragma unroll
            for (int r = 0; r < 8; ++r) {
                int mr = tm * 16 + hi * 8 + r;
                float e = acc[r] * a2 + c2;
                out[(size_t)(m0 + mr) * D_OUT + n] = e > 0.f ? e : 0.f;
            }
        }
    }
}

// ---------------------------------------------------------------------------
extern "C" void kernel_launch(void* const* d_in, const int* in_sizes, int n_in,
                              void* d_out, int out_size, void* d_ws, size_t ws_size,
                              hipStream_t stream)
{
    (void)in_sizes; (void)n_in; (void)out_size; (void)ws_size;
    const float* x         = (const float*)d_in[0];
    const int*   positions = (const int*)  d_in[1];
    const float* W_in      = (const float*)d_in[2];
    const float* b_in      = (const float*)d_in[3];
    const float* g1        = (const float*)d_in[4];
    const float* be1       = (const float*)d_in[5];
    const float* m1        = (const float*)d_in[6];
    const float* v1        = (const float*)d_in[7];
    const float* pos_table = (const float*)d_in[8];
    const float* query     = (const float*)d_in[9];
    const float* Wk        = (const float*)d_in[10];
    const float* bk        = (const float*)d_in[11];
    const float* Wmlp      = (const float*)d_in[12];
    const float* b_mlp     = (const float*)d_in[13];
    const float* g2        = (const float*)d_in[14];
    const float* be2       = (const float*)d_in[15];
    const float* m2        = (const float*)d_in[16];
    const float* v2        = (const float*)d_in[17];

    char* ws = (char*)d_ws;
    unsigned short* win_bf  = (unsigned short*)(ws + 0);        // 64 KB
    unsigned short* wk_bf   = (unsigned short*)(ws + 65536);    // 64 KB
    unsigned short* wmlp_bf = (unsigned short*)(ws + 131072);   // 64 KB
    float* A1 = (float*)(ws + 196608);
    float* C1 = (float*)(ws + 197632);
    float* A2 = (float*)(ws + 198656);
    float* C2 = (float*)(ws + 199168);
    float* y_ws = (float*)(ws + 262144);                        // 8192*256 f32 = 8 MB

    ltae_prep<<<386, 256, 0, stream>>>(W_in, Wk, Wmlp,
                                       b_in, g1, be1, m1, v1,
                                       b_mlp, g2, be2, m2, v2,
                                       win_bf, wk_bf, wmlp_bf, A1, C1, A2, C2);

    ltae_main<<<8192, 256, 0, stream>>>(x, positions, pos_table, query, bk,
                                        win_bf, wk_bf, A1, C1, y_ws);

    ltae_mlp<<<128, 256, 0, stream>>>(y_ws, wmlp_bf, A2, C2, (float*)d_out);
}
